// TemporalGraphNetwork_28604482191751
// MI455X (gfx1250) — matrix operand order repaired and checked by reading, not compiled, not run on hardware
//
#include <hip/hip_runtime.h>
#include <hip/hip_bf16.h>
#include <hip/hip_fp16.h>

typedef __attribute__((ext_vector_type(16))) _Float16 v16h;
typedef __attribute__((ext_vector_type(8)))  _Float16 v8h;
typedef __attribute__((ext_vector_type(8)))  float    v8f;

#define WMMA(a, b, c) __builtin_amdgcn_wmma_f32_16x16x32_f16( \
    false, (a), false, (b), (short)0, (c), false, false)

enum : int {
  NODE_F = 128, EDGE_F = 32, TIME_D = 16, MEM_D = 128, MSG_D = 128,
  MSG_IN = 176, MSG_K = 192,              // padded K for layer 1
  W1_FRAGS = 6 * 8,                       // 6 k-steps x 8 n-tiles
  W2_FRAGS = 4 * 8,
  GRU_FRAGS = 4 * 24,                     // 4 k-steps x 24 n-tiles (384 cols)
  OUT_FRAGS = 4 * 8,
  FRAG_HALFS = 32 * 16,                   // 32 lanes x 16 halfs
  WPB = 4                                 // waves per block
};

// ---- helpers ---------------------------------------------------------------

// Build an A-fragment (16x32 f16, ISA layout) from two runs of 8 floats.
// A layout per lane: h<8 -> K = 8*g + h ; h>=8 -> K = 16 + 8*g + (h-8).
static __device__ inline v16h cvt_a_frag(const float* p0, const float* p1) {
  v16h r;
#pragma unroll
  for (int j = 0; j < 8; ++j) r[j] = (_Float16)p0[j];
#pragma unroll
  for (int j = 0; j < 8; ++j) r[8 + j] = (_Float16)p1[j];
  return r;
}

// Load an A-fragment from f16 memory (two contiguous 16B runs -> b128 loads).
static __device__ inline v16h ld_a_frag(const _Float16* p0, const _Float16* p1) {
  v8h a = *(const v8h*)p0;
  v8h b = *(const v8h*)p1;
  return __builtin_shufflevector(a, b, 0, 1, 2, 3, 4, 5, 6, 7,
                                 8, 9, 10, 11, 12, 13, 14, 15);
}

static __device__ inline float fsigmoid(float x) {
  return 1.0f / (1.0f + __expf(-x));
}

// ---- kernel 0: pack weights into f16 WMMA B-fragments ----------------------
// B layout per fragment: lane l (g=l/16, n_lo=l%16) holds halfs h=0..15 with
//   K = ks*32 + (h<8 ? 8*g + h : 16 + 8*g + (h-8)),  N = nt*16 + n_lo.
__global__ void tgn_pack_weights(const float* __restrict__ msg_w1,   // [176,128]
                                 const float* __restrict__ msg_w2,   // [128,128]
                                 const float* __restrict__ gru_w_ih, // [384,128]
                                 const float* __restrict__ out_w,    // [128,128]
                                 _Float16* __restrict__ w1p,
                                 _Float16* __restrict__ w2p,
                                 _Float16* __restrict__ grup,
                                 _Float16* __restrict__ outp) {
  int tid = blockIdx.x * blockDim.x + threadIdx.x;
  int total = (W1_FRAGS + W2_FRAGS + GRU_FRAGS + OUT_FRAGS) * 32;
  if (tid >= total) return;
  int lane = tid & 31;
  int frag = tid >> 5;
  int g = lane >> 4, n_lo = lane & 15;

  if (frag < W1_FRAGS) {
    int ks = frag / 8, nt = frag % 8;
    _Float16* dst = w1p + (size_t)frag * FRAG_HALFS + lane * 16;
#pragma unroll
    for (int h = 0; h < 16; ++h) {
      int k = ks * 32 + (h < 8 ? g * 8 + h : 16 + g * 8 + (h - 8));
      int n = nt * 16 + n_lo;
      dst[h] = (_Float16)((k < MSG_IN) ? msg_w1[k * MSG_D + n] : 0.0f);
    }
  } else if (frag < W1_FRAGS + W2_FRAGS) {
    int f = frag - W1_FRAGS;
    int ks = f / 8, nt = f % 8;
    _Float16* dst = w2p + (size_t)f * FRAG_HALFS + lane * 16;
#pragma unroll
    for (int h = 0; h < 16; ++h) {
      int k = ks * 32 + (h < 8 ? g * 8 + h : 16 + g * 8 + (h - 8));
      int n = nt * 16 + n_lo;
      dst[h] = (_Float16)msg_w2[k * MEM_D + n];
    }
  } else if (frag < W1_FRAGS + W2_FRAGS + GRU_FRAGS) {
    int f = frag - W1_FRAGS - W2_FRAGS;
    int ks = f / 24, nct = f % 24;
    _Float16* dst = grup + (size_t)f * FRAG_HALFS + lane * 16;
#pragma unroll
    for (int h = 0; h < 16; ++h) {
      int k = ks * 32 + (h < 8 ? g * 8 + h : 16 + g * 8 + (h - 8));         // 0..127
      int n = nct * 16 + n_lo;              // 0..383
      dst[h] = (_Float16)gru_w_ih[n * MEM_D + k];   // B = W_ih^T
    }
  } else {
    int f = frag - W1_FRAGS - W2_FRAGS - GRU_FRAGS;
    int ks = f / 8, nt = f % 8;
    _Float16* dst = outp + (size_t)f * FRAG_HALFS + lane * 16;
#pragma unroll
    for (int h = 0; h < 16; ++h) {
      int k = ks * 32 + (h < 8 ? g * 8 + h : 16 + g * 8 + (h - 8));
      int n = nt * 16 + n_lo;
      dst[h] = (_Float16)out_w[n * MEM_D + k];      // B = out_w^T
    }
  }
}

// ---- kernel 1: zero agg + cnt ----------------------------------------------
__global__ void tgn_zero(float* __restrict__ p, int n) {
  int i = blockIdx.x * blockDim.x + threadIdx.x;
  if (i < n) p[i] = 0.0f;
}

// ---- kernel 2: edge message MLP + scatter-add ------------------------------
// launch_bounds min-waves=4: cap VGPRs at 256/wave so 4 waves/SIMD can hide
// L2-gather + atomic latency (natural live state is only ~120 VGPRs).
__global__ __launch_bounds__(WPB * 32, 4) void tgn_edge_kernel(
    const float* __restrict__ x,          // [N,128]
    const float* __restrict__ edge_attr,  // [E,32]
    const float* __restrict__ t,          // [E]
    const float* __restrict__ time_w,     // [16]
    const float* __restrict__ time_b,     // [16]
    const float* __restrict__ msg_b1,     // [128]
    const float* __restrict__ msg_b2,     // [128]
    const int* __restrict__ edge_index,   // [2,E]
    const _Float16* __restrict__ w1p_g,   // packed layer-1 B frags
    const _Float16* __restrict__ w2p_g,   // packed layer-2 B frags
    float* __restrict__ agg,              // [N,128] atomic accum
    float* __restrict__ cnt,              // [N] atomic count
    int n_edges) {
  __shared__ _Float16 w1s[W1_FRAGS * FRAG_HALFS];      // 48 KB
  __shared__ _Float16 w2s[W2_FRAGS * FRAG_HALFS];      // 32 KB
  __shared__ _Float16 hbuf[WPB][16 * MSG_D];           // 16 KB transpose scratch

  // cooperative stage of both weight-fragment buffers into LDS (320 KB/WGP)
  for (int i = threadIdx.x; i < (W1_FRAGS * FRAG_HALFS) / 8; i += blockDim.x)
    ((v8h*)w1s)[i] = ((const v8h*)w1p_g)[i];
  for (int i = threadIdx.x; i < (W2_FRAGS * FRAG_HALFS) / 8; i += blockDim.x)
    ((v8h*)w2s)[i] = ((const v8h*)w2p_g)[i];
  __syncthreads();

  const int lane = threadIdx.x & 31;
  const int w = threadIdx.x >> 5;
  const int g = lane >> 4;
  const int m = lane & 15;
  const int n_tiles = n_edges >> 4;
  _Float16* hb = hbuf[w];

  for (int tile = blockIdx.x * WPB + w; tile < n_tiles;
       tile += gridDim.x * WPB) {
    const int e = tile * 16 + m;                 // this lane's A-row edge
    const int src = edge_index[e];

    // ---- build A fragments of msg_in (16 edges x 192 cols, K-padded) ----
    v16h A1[6];
    const float* xr = x + (size_t)src * NODE_F;
#pragma unroll
    for (int ks = 0; ks < 4; ++ks)
      A1[ks] = cvt_a_frag(xr + ks * 32 + g * 8, xr + ks * 32 + 16 + g * 8);
    {
      const float* ea = edge_attr + (size_t)e * EDGE_F;
      A1[4] = cvt_a_frag(ea + g * 8, ea + 16 + g * 8);
    }
    {
      float tv = t[e];
      v16h a;
#pragma unroll
      for (int j = 0; j < 8; ++j) {
        int k = g * 8 + j;
        a[j] = (_Float16)__cosf(tv * time_w[k] + time_b[k]);
      }
#pragma unroll
      for (int j = 8; j < 16; ++j) a[j] = (_Float16)0.0f;   // K pad 176..191
      A1[5] = a;
    }

    // ---- layer 1: [16,192] @ [192,128], ReLU, transpose via LDS ----
#pragma unroll
    for (int nt = 0; nt < 8; ++nt) {
      float bias = msg_b1[nt * 16 + m];
      v8f c;
#pragma unroll
      for (int v = 0; v < 8; ++v) c[v] = bias;
#pragma unroll
      for (int ks = 0; ks < 6; ++ks) {
        v16h b = *(const v16h*)(w1s + ((ks * 8 + nt) * 32 + lane) * 16);
        c = WMMA(A1[ks], b, c);
      }
#pragma unroll
      for (int v = 0; v < 8; ++v) {
        float r = c[v] > 0.0f ? c[v] : 0.0f;
        hb[(v + 8 * g) * MSG_D + nt * 16 + m] = (_Float16)r;  // [row][col]
      }
    }

    // ---- reload h1 in A layout (per-wave LDS, DS ops in-order) ----
    v16h A2[4];
#pragma unroll
    for (int k2 = 0; k2 < 4; ++k2)
      A2[k2] = ld_a_frag(hb + m * MSG_D + k2 * 32 + g * 8,
                         hb + m * MSG_D + k2 * 32 + 16 + g * 8);

    int tgt8[8];
#pragma unroll
    for (int v = 0; v < 8; ++v)
      tgt8[v] = edge_index[n_edges + tile * 16 + v + 8 * g];

    // ---- layer 2: [16,128] @ [128,128] + scatter atomics ----
#pragma unroll
    for (int nt = 0; nt < 8; ++nt) {
      float bias = msg_b2[nt * 16 + m];
      v8f c;
#pragma unroll
      for (int v = 0; v < 8; ++v) c[v] = bias;
#pragma unroll
      for (int k2 = 0; k2 < 4; ++k2) {
        v16h b = *(const v16h*)(w2s + ((k2 * 8 + nt) * 32 + lane) * 16);
        c = WMMA(A2[k2], b, c);
      }
#pragma unroll
      for (int v = 0; v < 8; ++v)
        unsafeAtomicAdd(&agg[(size_t)tgt8[v] * MEM_D + nt * 16 + m], c[v]);
    }

    if (lane < 16) {
      int tg = edge_index[n_edges + e];
      unsafeAtomicAdd(&cnt[tg], 1.0f);
    }
  }
}

// ---- kernel 3: mean + GRU gates + output projection ------------------------
__global__ __launch_bounds__(WPB * 32, 4) void tgn_node_kernel(
    const float* __restrict__ agg,        // [N,128]
    const float* __restrict__ cnt,        // [N]
    const _Float16* __restrict__ grup,    // packed W_ih^T frags [4][24]
    const _Float16* __restrict__ outp,    // packed out_w^T frags [4][8]
    const float* __restrict__ gru_b_ih,   // [384]
    const float* __restrict__ gru_b_hh,   // [384]
    const float* __restrict__ out_b,      // [128]
    float* __restrict__ out,              // [N,128]
    int n_nodes) {
  __shared__ _Float16 hbuf[WPB][16 * MEM_D];   // 16 KB transpose scratch

  const int lane = threadIdx.x & 31;
  const int w = threadIdx.x >> 5;
  const int g = lane >> 4;
  const int m = lane & 15;
  const int n_tiles = n_nodes >> 4;            // 50000 / 16 = 3125
  _Float16* hb = hbuf[w];

  for (int tile = blockIdx.x * WPB + w; tile < n_tiles;
       tile += gridDim.x * WPB) {
    const int row = tile * 16 + m;
    const float inv = 1.0f / fmaxf(cnt[row], 1.0f);
    const float* ar = agg + (size_t)row * MEM_D;

    // A fragments = mean-normalized aggregate, f16
    v16h A[4];
#pragma unroll
    for (int k2 = 0; k2 < 4; ++k2) {
      v16h a;
#pragma unroll
      for (int j = 0; j < 8; ++j) a[j] = (_Float16)(ar[k2 * 32 + g * 8 + j] * inv);
#pragma unroll
      for (int j = 0; j < 8; ++j)
        a[8 + j] = (_Float16)(ar[k2 * 32 + 16 + g * 8 + j] * inv);
      A[k2] = a;
    }

    // GRU: gi = agg @ W_ih^T + b_ih ; gh = b_hh (h0 == 0)
#pragma unroll
    for (int nt = 0; nt < 8; ++nt) {
      const int col = nt * 16 + m;
      v8f cr, cz, cn;
      float br = gru_b_ih[col], bz = gru_b_ih[128 + col], bn = gru_b_ih[256 + col];
#pragma unroll
      for (int v = 0; v < 8; ++v) { cr[v] = br; cz[v] = bz; cn[v] = bn; }
#pragma unroll
      for (int k2 = 0; k2 < 4; ++k2) {
        v16h b_r = *(const v16h*)(grup + ((size_t)(k2 * 24 + nt) * 32 + lane) * 16);
        v16h b_z = *(const v16h*)(grup + ((size_t)(k2 * 24 + nt + 8) * 32 + lane) * 16);
        v16h b_n = *(const v16h*)(grup + ((size_t)(k2 * 24 + nt + 16) * 32 + lane) * 16);
        cr = WMMA(A[k2], b_r, cr);
        cz = WMMA(A[k2], b_z, cz);
        cn = WMMA(A[k2], b_n, cn);
      }
      float bhr = gru_b_hh[col], bhz = gru_b_hh[128 + col], bhn = gru_b_hh[256 + col];
#pragma unroll
      for (int v = 0; v < 8; ++v) {
        float r = fsigmoid(cr[v] + bhr);
        float z = fsigmoid(cz[v] + bhz);
        float n = tanhf(cn[v] + r * bhn);
        float mem = (1.0f - z) * n;                       // + z*h0, h0 == 0
        hb[(v + 8 * g) * MEM_D + col] = (_Float16)mem;
      }
    }

    // reload memory in A layout, output projection
    v16h A3[4];
#pragma unroll
    for (int k2 = 0; k2 < 4; ++k2)
      A3[k2] = ld_a_frag(hb + m * MEM_D + k2 * 32 + g * 8,
                         hb + m * MEM_D + k2 * 32 + 16 + g * 8);

#pragma unroll
    for (int nt = 0; nt < 8; ++nt) {
      float bias = out_b[nt * 16 + m];
      v8f c;
#pragma unroll
      for (int v = 0; v < 8; ++v) c[v] = bias;
#pragma unroll
      for (int k2 = 0; k2 < 4; ++k2) {
        v16h b = *(const v16h*)(outp + ((size_t)(k2 * 8 + nt) * 32 + lane) * 16);
        c = WMMA(A3[k2], b, c);
      }
#pragma unroll
      for (int v = 0; v < 8; ++v)
        out[(size_t)(tile * 16 + v + 8 * g) * NODE_F + nt * 16 + m] = c[v];
    }
  }
}

// ---- host launch -----------------------------------------------------------
extern "C" void kernel_launch(void* const* d_in, const int* in_sizes, int n_in,
                              void* d_out, int out_size, void* d_ws, size_t ws_size,
                              hipStream_t stream) {
  const float* x         = (const float*)d_in[0];
  const float* edge_attr = (const float*)d_in[1];
  const float* t         = (const float*)d_in[2];
  const float* time_w    = (const float*)d_in[3];
  const float* time_b    = (const float*)d_in[4];
  const float* msg_w1    = (const float*)d_in[5];
  const float* msg_b1    = (const float*)d_in[6];
  const float* msg_w2    = (const float*)d_in[7];
  const float* msg_b2    = (const float*)d_in[8];
  const float* gru_w_ih  = (const float*)d_in[9];
  // d_in[10] = gru_w_hh (unused: h0 == 0)
  const float* gru_b_ih  = (const float*)d_in[11];
  const float* gru_b_hh  = (const float*)d_in[12];
  const float* out_w     = (const float*)d_in[13];
  const float* out_b     = (const float*)d_in[14];
  const int*   edge_index = (const int*)d_in[15];

  const int n_edges = in_sizes[2];
  const int n_nodes = in_sizes[0] / NODE_F;

  // workspace carve-up (bytes)
  char* ws = (char*)d_ws;
  size_t off = 0;
  float* agg = (float*)(ws + off); off += (size_t)n_nodes * MEM_D * sizeof(float);
  float* cnt = (float*)(ws + off); off += (size_t)n_nodes * sizeof(float);
  off = (off + 255) & ~(size_t)255;
  _Float16* w1p  = (_Float16*)(ws + off); off += (size_t)W1_FRAGS  * FRAG_HALFS * 2;
  _Float16* w2p  = (_Float16*)(ws + off); off += (size_t)W2_FRAGS  * FRAG_HALFS * 2;
  _Float16* grup = (_Float16*)(ws + off); off += (size_t)GRU_FRAGS * FRAG_HALFS * 2;
  _Float16* outp = (_Float16*)(ws + off); off += (size_t)OUT_FRAGS * FRAG_HALFS * 2;

  // 0) pack weights into WMMA B fragments
  {
    int threads = (W1_FRAGS + W2_FRAGS + GRU_FRAGS + OUT_FRAGS) * 32;
    int blk = 256;
    tgn_pack_weights<<<(threads + blk - 1) / blk, blk, 0, stream>>>(
        msg_w1, msg_w2, gru_w_ih, out_w, w1p, w2p, grup, outp);
  }
  // 1) zero accumulators
  {
    int n = n_nodes * (MEM_D + 1);
    tgn_zero<<<(n + 255) / 256, 256, 0, stream>>>(agg, n);
  }
  // 2) edge message MLP + scatter
  tgn_edge_kernel<<<1024, WPB * 32, 0, stream>>>(
      x, edge_attr, t, time_w, time_b, msg_b1, msg_b2, edge_index,
      w1p, w2p, agg, cnt, n_edges);
  // 3) mean + GRU + output projection
  tgn_node_kernel<<<256, WPB * 32, 0, stream>>>(
      agg, cnt, grup, outp, gru_b_ih, gru_b_hh, out_b,
      (float*)d_out, n_nodes);
}
